// RBFN_57715770523862
// MI455X (gfx1250) — compile-verified, matrix operand order checked by reading, not run on hardware
//
#include <hip/hip_runtime.h>
#include <math.h>

// Problem constants (from reference setup_inputs)
#define BB 1024      // batch
#define MM 209       // centers
#define DD 12288     // feature dim
#define SS 2         // outputs
#define MP 224       // M padded to 14 * 16
#define NTILES 14    // real 16-wide N tiles (224/16)
#define BLK_ROWS 64  // output rows per block (4 row-subtiles)
#define RSUB 4       // row-subtiles per block
#define KCH 32       // K chunk per WMMA step
#define NSPLIT 8     // deterministic split-K factor
#define KSPLIT (DD / NSPLIT)   // 1536
#define NIT (KSPLIT / KCH)     // 48 chunks per block
#define LDSROW 80    // bytes per A row in LDS (64B data + 16B pad: conflict-free, 16B aligned)

typedef __attribute__((ext_vector_type(16))) __bf16 v16bf;
typedef __attribute__((ext_vector_type(8)))  __bf16 v8bf;
typedef __attribute__((ext_vector_type(8)))  float  v8f;
typedef __attribute__((ext_vector_type(4)))  float  v4f;

// ---------------------------------------------------------------------------
// Kernel 1: row-wise sum of squares for x (1024 rows) and center (209 rows).
// One block per row; b128 streaming pass (also warms the 192MB L2 for the GEMM).
// ---------------------------------------------------------------------------
__global__ __launch_bounds__(256) void rbf_rowsq(const float* __restrict__ x,
                                                 const float* __restrict__ c,
                                                 float* __restrict__ x2,
                                                 float* __restrict__ c2) {
  const int b = blockIdx.x;
  const float* __restrict__ src;
  float* __restrict__ dst;
  if (b < BB) {
    src = x + (size_t)b * DD;
    dst = x2 + b;
  } else {
    src = c + (size_t)(b - BB) * DD;
    dst = c2 + (b - BB);
  }
  const int t = threadIdx.x;
  float s = 0.0f;
  const v4f* __restrict__ src4 = (const v4f*)src;
  for (int i = t; i < DD / 4; i += 256) {
    v4f v = src4[i];
    s = fmaf(v[0], v[0], s);
    s = fmaf(v[1], v[1], s);
    s = fmaf(v[2], v[2], s);
    s = fmaf(v[3], v[3], s);
  }
  __shared__ float red[256];
  red[t] = s;
  __syncthreads();
  for (int off = 128; off > 0; off >>= 1) {
    if (t < off) red[t] += red[t + off];
    __syncthreads();
  }
  if (t == 0) *dst = red[0];
}

// ---------------------------------------------------------------------------
// Kernel 2: split-K bf16 WMMA GEMM, f32 accumulate.
//   Grid: (16 row-blocks, 8 K-splits) = 128 blocks x 8 waves.
//   Block tile: 64 rows x 224(+pad) cols. Wave w owns N-tiles (2w, 2w+1) and
//   all 4 row-subtiles -> 8 accumulators, 8 WMMA per K-chunk.
//   A chunk (64x32 f32) is converted to bf16 once by the 256 threads and
//   double-buffered in LDS (80B row stride -> conflict-free ds_load_b128).
//   B is streamed per-wave from global (L2 resident), with global_prefetch_b8
//   two chunks ahead.
//
// WMMA operand layout (ISA 7.12.2, 16-bit A 16x32): lane -> M row (mod 16),
//   h = lane>>4; elements 0..7 -> K = h*8+j, elements 8..15 -> K = 16+h*8+j.
// B fed with the symmetric per-lane layout (lane = N column = one center row).
// C/D: element i, lane L -> M = i + 8*(L>>4), N = L&15.
// ---------------------------------------------------------------------------
__global__ __launch_bounds__(256) void rbf_gemm(const float* __restrict__ x,
                                                const float* __restrict__ c,
                                                float* __restrict__ dotp) {
  __shared__ __align__(16) unsigned char ldsA[2][BLK_ROWS * LDSROW];

  const int tid  = threadIdx.x;
  const int wave = tid >> 5;      // 0..7
  const int lane = tid & 31;
  const int m16  = lane & 15;
  const int h    = lane >> 4;

  const int rowBase = blockIdx.x * BLK_ROWS;
  const int kBase   = blockIdx.y * KSPLIT;

  // A-stage loader: thread t stages 8 floats (2x b128 load, 1x ds_store_b128)
  const int ldRow = tid >> 2;        // 0..63
  const int ldOff = (tid & 3) * 8;   // 0,8,16,24
  const float* __restrict__ aSrc =
      x + (size_t)(rowBase + ldRow) * DD + kBase + ldOff;
  const int ldsOff = ldRow * LDSROW + (tid & 3) * 16;

  // B rows (centers) for the two N-tiles owned by this wave; clamp padding
  const int n0 = wave * 2, n1 = n0 + 1;          // tiles 0..15 (14,15 = pad)
  const int col0 = n0 * 16 + m16, col1 = n1 * 16 + m16;
  const int cc0 = col0 < MM ? col0 : MM - 1;
  const int cc1 = col1 < MM ? col1 : MM - 1;
  const float* __restrict__ bRow0 = c + (size_t)cc0 * DD + kBase;
  const float* __restrict__ bRow1 = c + (size_t)cc1 * DD + kBase;

  v8f acc[RSUB][2];
#pragma unroll
  for (int s = 0; s < RSUB; ++s) {
    acc[s][0] = {};
    acc[s][1] = {};
  }

  // stage chunk 0
  {
    const v4f* pa = (const v4f*)aSrc;
    v4f v0 = pa[0], v1 = pa[1];
    v8bf w;
#pragma unroll
    for (int j = 0; j < 4; ++j) {
      w[j]     = (__bf16)v0[j];
      w[4 + j] = (__bf16)v1[j];
    }
    *(v8bf*)&ldsA[0][ldsOff] = w;
  }
  __syncthreads();

  for (int it = 0; it < NIT; ++it) {
    const int kOff = it * KCH;
    const unsigned char* curBase = (it & 1) ? &ldsA[1][0] : &ldsA[0][0];
    unsigned char* nxtBase = (it & 1) ? &ldsA[0][0] : &ldsA[1][0];

    // stage next A chunk into the other LDS buffer (f32 -> bf16 once)
    if (it + 1 < NIT) {
      const v4f* pa = (const v4f*)(aSrc + kOff + KCH);
      v4f v0 = pa[0], v1 = pa[1];
      v8bf w;
#pragma unroll
      for (int j = 0; j < 4; ++j) {
        w[j]     = (__bf16)v0[j];
        w[4 + j] = (__bf16)v1[j];
      }
      *(v8bf*)&nxtBase[ldsOff] = w;
    }

    // prefetch the B stream two chunks ahead (global_prefetch_b8)
    __builtin_prefetch(bRow0 + kOff + 2 * KCH, 0, 1);
    __builtin_prefetch(bRow1 + kOff + 2 * KCH, 0, 1);

    // B operands from global, convert f32 -> bf16 in registers
    const int base = kOff + h * 8;
    const v4f* pb0 = (const v4f*)(bRow0 + base);
    v4f b0a = pb0[0], b0b = pb0[1], b0c = pb0[4], b0d = pb0[5];
    const v4f* pb1 = (const v4f*)(bRow1 + base);
    v4f b1a = pb1[0], b1b = pb1[1], b1c = pb1[4], b1d = pb1[5];
    v16bf bm0, bm1;
#pragma unroll
    for (int j = 0; j < 4; ++j) {
      bm0[j]      = (__bf16)b0a[j];
      bm0[4 + j]  = (__bf16)b0b[j];
      bm0[8 + j]  = (__bf16)b0c[j];
      bm0[12 + j] = (__bf16)b0d[j];
      bm1[j]      = (__bf16)b1a[j];
      bm1[4 + j]  = (__bf16)b1b[j];
      bm1[8 + j]  = (__bf16)b1c[j];
      bm1[12 + j] = (__bf16)b1d[j];
    }

    // 4 row-subtiles of A from LDS, 8 WMMAs (A reused for both N-tiles)
#pragma unroll
    for (int s = 0; s < RSUB; ++s) {
      const unsigned char* l = curBase + (s * 16 + m16) * LDSROW + h * 16;
      v8bf alo = *(const v8bf*)l;
      v8bf ahi = *(const v8bf*)(l + 32);
      v16bf a = __builtin_shufflevector(alo, ahi, 0, 1, 2, 3, 4, 5, 6, 7, 8, 9,
                                        10, 11, 12, 13, 14, 15);
      acc[s][0] = __builtin_amdgcn_wmma_f32_16x16x32_bf16(
          false, a, false, bm0, (short)0, acc[s][0], false, false);
      acc[s][1] = __builtin_amdgcn_wmma_f32_16x16x32_bf16(
          false, a, false, bm1, (short)0, acc[s][1], false, false);
    }

    __syncthreads();  // next chunk visible; everyone done reading curBase
  }

  // store this K-split's partial tile (exclusive ownership -> deterministic)
  float* __restrict__ part = dotp + (size_t)blockIdx.y * (BB * MP);
#pragma unroll
  for (int s = 0; s < RSUB; ++s) {
#pragma unroll
    for (int i = 0; i < 8; ++i) {
      const int r = rowBase + s * 16 + 8 * h + i;
      if (n0 < NTILES) part[(size_t)r * MP + n0 * 16 + m16] = acc[s][0][i];
      if (n1 < NTILES) part[(size_t)r * MP + n1 * 16 + m16] = acc[s][1][i];
    }
  }
}

// ---------------------------------------------------------------------------
// Kernel 3: epilogue. One block per batch row.
//   dot = sum of NSPLIT partials (fixed order); d2 = x2 + c2 - 2*dot;
//   radial = exp(-sqrt(max(d2,0))/sigma^2); out[b,s] = radial . W[s,:] + bias
// ---------------------------------------------------------------------------
__global__ __launch_bounds__(256) void rbf_epilogue(
    const float* __restrict__ dotp, const float* __restrict__ x2,
    const float* __restrict__ c2, const float* __restrict__ sigma,
    const float* __restrict__ W, const float* __restrict__ bias,
    float* __restrict__ out) {
  const int row = blockIdx.x;
  const int t   = threadIdx.x;

  float s0 = 0.0f, s1 = 0.0f;
  if (t < MM) {
    float dot = 0.0f;
#pragma unroll
    for (int p = 0; p < NSPLIT; ++p)
      dot += dotp[(size_t)p * (BB * MP) + (size_t)row * MP + t];
    const float d2     = x2[row] + c2[t] - 2.0f * dot;
    const float dist   = sqrtf(fmaxf(d2, 0.0f));
    const float sg     = sigma[t];
    const float radial = expf(-dist / (sg * sg));
    s0 = radial * W[t];        // W[0][m]
    s1 = radial * W[MM + t];   // W[1][m]
  }

  __shared__ float red0[256];
  __shared__ float red1[256];
  red0[t] = s0;
  red1[t] = s1;
  __syncthreads();
  for (int off = 128; off > 0; off >>= 1) {
    if (t < off) {
      red0[t] += red0[t + off];
      red1[t] += red1[t + off];
    }
    __syncthreads();
  }
  if (t == 0) {
    out[row * SS + 0] = red0[0] + bias[0];
    out[row * SS + 1] = red1[0] + bias[1];
  }
}

// ---------------------------------------------------------------------------
// Launch: rowsq -> gemm (split-K x8) -> epilogue, all on `stream`.
// Workspace: dot partials [8][1024][224] f32 (7.34MB) | x2[1024] | c2[209]
// ---------------------------------------------------------------------------
extern "C" void kernel_launch(void* const* d_in, const int* in_sizes, int n_in,
                              void* d_out, int out_size, void* d_ws,
                              size_t ws_size, hipStream_t stream) {
  const float* x     = (const float*)d_in[0];  // [1024,12288]
  const float* c     = (const float*)d_in[1];  // [209,12288]
  const float* sigma = (const float*)d_in[2];  // [209]
  const float* W     = (const float*)d_in[3];  // [2,209]
  const float* bias  = (const float*)d_in[4];  // [2]
  float* out = (float*)d_out;                  // [1024,2]

  float* dotp = (float*)d_ws;                         // NSPLIT * BB * MP
  float* x2   = dotp + (size_t)NSPLIT * BB * MP;
  float* c2   = x2 + BB;

  rbf_rowsq<<<BB + MM, 256, 0, stream>>>(x, c, x2, c2);
  rbf_gemm<<<dim3(BB / BLK_ROWS, NSPLIT), 256, 0, stream>>>(x, c, dotp);
  rbf_epilogue<<<BB, 256, 0, stream>>>(dotp, x2, c2, sigma, W, bias, out);
}
